// MHSA_81612968559005
// MI455X (gfx1250) — matrix operand order repaired and checked by reading, not compiled, not run on hardware
//
#include <hip/hip_runtime.h>
#include <stdint.h>

// ---------- types ----------
typedef __attribute__((ext_vector_type(16))) __bf16        v16bf;
typedef __attribute__((ext_vector_type(8)))  float         v8f;
typedef __attribute__((ext_vector_type(4)))  unsigned int  u32x4;
typedef __attribute__((ext_vector_type(8)))  int           i32x8;
typedef __attribute__((ext_vector_type(4)))  int           i32x4;

union Frag16 {
  v16bf v;
  u32x4 q[2];
  unsigned short h[16];
};

static __device__ __forceinline__ unsigned short f2bf(float f) {
  unsigned u = __builtin_bit_cast(unsigned, f);
  unsigned r = u + 0x7FFFu + ((u >> 16) & 1u);   // round-to-nearest-even
  return (unsigned short)(r >> 16);
}

static __device__ __forceinline__ v8f wmma_bf16(v16bf a, v16bf b, v8f c) {
  // D(f32 16x16) = A(bf16 16x32) * B(bf16 32x16) + C
  return __builtin_amdgcn_wmma_f32_16x16x32_bf16(false, a, false, b, (short)0, c,
                                                 false, false);
}

static __device__ __forceinline__ float red_max16(float v) {
  v = fmaxf(v, __shfl_xor(v, 1, 32));
  v = fmaxf(v, __shfl_xor(v, 2, 32));
  v = fmaxf(v, __shfl_xor(v, 4, 32));
  v = fmaxf(v, __shfl_xor(v, 8, 32));
  return v;
}
static __device__ __forceinline__ float red_sum16(float v) {
  v += __shfl_xor(v, 1, 32);
  v += __shfl_xor(v, 2, 32);
  v += __shfl_xor(v, 4, 32);
  v += __shfl_xor(v, 8, 32);
  return v;
}

// ---------- TDM: async DMA of a (tileH x tileW) bf16 tile into LDS ----------
static __device__ __forceinline__ void tdm_load_bf16(const void* gptr, unsigned ldsoff,
                                                     int tileW, int tileH,
                                                     int strideElems) {
  unsigned long long ga = (unsigned long long)(uintptr_t)gptr;
  u32x4 g0;
  g0[0] = 1u;                                                  // count=1, no gather
  g0[1] = ldsoff;                                              // lds_addr (bytes)
  g0[2] = (unsigned)ga;                                        // global_addr[31:0]
  g0[3] = ((unsigned)(ga >> 32) & 0x01FFFFFFu) | (2u << 30);   // addr[56:32] | type=2
  i32x8 g1;
  g1[0] = 1 << 16;        // data_size = 2 bytes
  g1[1] = tileW << 16;    // tensor_dim0
  g1[2] = tileH << 16;    // tensor_dim1
  g1[3] = tileW << 16;    // tile_dim0
  g1[4] = tileH;          // tile_dim1, tile_dim2 = 0
  g1[5] = strideElems;    // tensor_dim0_stride
  g1[6] = 0;
  g1[7] = 0;
  i32x4 z4 = {0, 0, 0, 0};
#if __has_include(<hip/amd_detail/amd_gfx1250_TDM.h>)
  i32x8 z8 = {0, 0, 0, 0, 0, 0, 0, 0};
  __builtin_amdgcn_tensor_load_to_lds(g0, g1, z4, z4, z8, 0);
#else
  __builtin_amdgcn_tensor_load_to_lds(g0, g1, z4, z4, 0);
#endif
}

// ---------- elementwise casts ----------
__global__ void cast_bf(const float* __restrict__ in, unsigned short* __restrict__ out,
                        int n) {
  int i = blockIdx.x * blockDim.x + threadIdx.x;
  if (i < n) out[i] = f2bf(in[i]);
}

// in: K x N row-major (fp32) -> out: N x K row-major (bf16), i.e. transpose+cast
__global__ void tcast_bf(const float* __restrict__ in, unsigned short* __restrict__ out,
                         int K, int N) {
  int n = blockIdx.x * 16 + threadIdx.x;
  int k = blockIdx.y * 16 + threadIdx.y;
  out[(size_t)n * K + k] = f2bf(in[(size_t)k * N + n]);
}

// ---------- V transpose: qkv V slice [b*2048][1024] (stride 3072) -> vT [b][c][seq] ----------
// vT[(bi*1024 + c)*2048 + n] = qkv[(bi*2048 + n)*3072 + 2048 + c]; LDS tile, both sides coalesced
__global__ void __launch_bounds__(256)
vtrans_bf(const unsigned short* __restrict__ qkv, unsigned short* __restrict__ vT) {
  __shared__ unsigned short tile[16][17];
  const int bi = blockIdx.z;
  const int n0 = blockIdx.x * 16;
  const int c0 = blockIdx.y * 16;
  const int x = threadIdx.x, y = threadIdx.y;
  tile[y][x] = qkv[(size_t)(bi * 2048 + n0 + y) * 3072 + 2048 + c0 + x];
  __syncthreads();
  vT[(size_t)(bi * 1024 + c0 + y) * 2048 + n0 + x] = tile[x][y];
}

// ---------- GEMM: C[M x N] = A[M x K](bf16) * Bt[N x K](bf16)^T + bias ----------
// grid = (N/128, M/128), block = 256 (8 waves). TDM double-buffers 128x32 A/B tiles.
__global__ void __launch_bounds__(256)
gemm_bf16(const unsigned short* __restrict__ A, const unsigned short* __restrict__ Bt,
          const float* __restrict__ bias, void* __restrict__ Cout,
          int N, int K, int outIsBf16) {
  __shared__ alignas(16) unsigned short As[2][128 * 32];  // [m][k] packed rows of 32
  __shared__ alignas(16) unsigned short Bs[2][128 * 32];  // [n][k] packed rows of 32

  const int wave = threadIdx.x >> 5;
  const int lane = threadIdx.x & 31;
  const int r  = lane & 15;
  const int hf = lane >> 4;
  const int mb = blockIdx.y * 128;
  const int nb = blockIdx.x * 128;

  const unsigned short* Ag = A  + (size_t)mb * K;
  const unsigned short* Bg = Bt + (size_t)nb * K;
  const int nsteps = K >> 5;

  if (wave == 0) {  // prefetch k-step 0
    tdm_load_bf16(Ag, (unsigned)(uintptr_t)&As[0][0], 32, 128, K);
    tdm_load_bf16(Bg, (unsigned)(uintptr_t)&Bs[0][0], 32, 128, K);
  }

  v8f c[8];
#pragma unroll
  for (int f = 0; f < 8; ++f) c[f] = (v8f){0.f, 0.f, 0.f, 0.f, 0.f, 0.f, 0.f, 0.f};

  for (int s = 0; s < nsteps; ++s) {
    const int buf = s & 1;
    if (wave == 0) {
      if (s + 1 < nsteps) {
        tdm_load_bf16(Ag + (s + 1) * 32, (unsigned)(uintptr_t)&As[buf ^ 1][0], 32, 128, K);
        tdm_load_bf16(Bg + (s + 1) * 32, (unsigned)(uintptr_t)&Bs[buf ^ 1][0], 32, 128, K);
        __builtin_amdgcn_s_wait_tensorcnt(2);
      } else {
        __builtin_amdgcn_s_wait_tensorcnt(0);
      }
    }
    __syncthreads();

    const unsigned short* As_ = &As[buf][0];
    const unsigned short* Bs_ = &Bs[buf][0];

    Frag16 a;
    const unsigned short* ap = As_ + (wave * 16 + r) * 32;
    a.q[0] = *(const u32x4*)(ap + 16 * hf);
    a.q[1] = *(const u32x4*)(ap + 16 * hf + 8);
    Frag16 b[8];
#pragma unroll
    for (int f = 0; f < 8; ++f) {
      const unsigned short* bp = Bs_ + (f * 16 + r) * 32 + 16 * hf;
      b[f].q[0] = *(const u32x4*)bp;
      b[f].q[1] = *(const u32x4*)(bp + 8);
    }
#pragma unroll
    for (int f = 0; f < 8; ++f) c[f] = wmma_bf16(a.v, b[f].v, c[f]);

    __syncthreads();
  }

#pragma unroll
  for (int f = 0; f < 8; ++f) {
    const int col = nb + f * 16 + r;
    const float bs = bias[col];
#pragma unroll
    for (int j = 0; j < 8; ++j) {
      const size_t row = (size_t)(mb + wave * 16) + j + 8 * hf;
      const float v = c[f][j] + bs;
      if (outIsBf16) ((unsigned short*)Cout)[row * N + col] = f2bf(v);
      else           ((float*)Cout)[row * N + col] = v;
    }
  }
}

// ---------- fused flash attention ----------
// qkv: [B*2048][3072] bf16; vT: [B*1024][2048] bf16 (V transposed); out: [B*2048][1024] bf16
#define SEQ 2048
#define QKV_LD 3072
#define NKT 64   // 2048 / 32 key tiles

__global__ void __launch_bounds__(256)
attn_kernel(const unsigned short* __restrict__ qkv, const unsigned short* __restrict__ vT,
            unsigned short* __restrict__ outbf) {
  __shared__ alignas(16) unsigned short Kt[2][32 * 64];  // key tile, row-major [key][hd]
  __shared__ alignas(16) unsigned short Vt[2][64 * 32];  // V^T tile, [hd][key]
  __shared__ alignas(16) unsigned short Pb[8][16 * 32];  // per-wave P transpose scratch

  const int bi   = blockIdx.z;
  const int head = blockIdx.y;
  const int wave = threadIdx.x >> 5;
  const int lane = threadIdx.x & 31;
  const int r  = lane & 15;
  const int hf = lane >> 4;

  const size_t rowbase = (size_t)bi * SEQ;
  const int qb = blockIdx.x * 128 + wave * 16;

  // Q fragments (A layout), resident for the whole key loop
  const unsigned short* qrow = qkv + (rowbase + qb + r) * QKV_LD + head * 64;
  Frag16 aq0, aq1;
  aq0.q[0] = *(const u32x4*)(qrow + 16 * hf);
  aq0.q[1] = *(const u32x4*)(qrow + 16 * hf + 8);
  aq1.q[0] = *(const u32x4*)(qrow + 32 + 16 * hf);
  aq1.q[1] = *(const u32x4*)(qrow + 40 + 16 * hf);

  v8f o[4];
#pragma unroll
  for (int f = 0; f < 4; ++f) o[f] = (v8f){0.f, 0.f, 0.f, 0.f, 0.f, 0.f, 0.f, 0.f};
  float mrun[8], lrun[8];
#pragma unroll
  for (int j = 0; j < 8; ++j) { mrun[j] = -3.0e38f; lrun[j] = 0.f; }

  const unsigned short* kbase = qkv + rowbase * QKV_LD + 1024 + head * 64;
  const unsigned short* vbase = vT + (size_t)(bi * 1024 + head * 64) * SEQ;  // [hd][seq]

  if (wave == 0) {  // prefetch tile 0 via Tensor Data Mover
    tdm_load_bf16(kbase, (unsigned)(uintptr_t)&Kt[0][0], 64, 32, QKV_LD);
    tdm_load_bf16(vbase, (unsigned)(uintptr_t)&Vt[0][0], 32, 64, SEQ);
  }

  for (int t = 0; t < NKT; ++t) {
    const int buf = t & 1;
    if (wave == 0) {
      if (t + 1 < NKT) {  // prefetch next tile into other buffer, then wait current
        tdm_load_bf16(kbase + (size_t)(t + 1) * 32 * QKV_LD,
                      (unsigned)(uintptr_t)&Kt[buf ^ 1][0], 64, 32, QKV_LD);
        tdm_load_bf16(vbase + (t + 1) * 32,
                      (unsigned)(uintptr_t)&Vt[buf ^ 1][0], 32, 64, SEQ);
        __builtin_amdgcn_s_wait_tensorcnt(2);
      } else {
        __builtin_amdgcn_s_wait_tensorcnt(0);
      }
    }
    __syncthreads();

    const unsigned short* Kt_ = &Kt[buf][0];
    const unsigned short* Vt_ = &Vt[buf][0];

    // S = Q * K^T : row-major K tile is exactly the column-major B operand
    v8f cL = (v8f){0.f, 0.f, 0.f, 0.f, 0.f, 0.f, 0.f, 0.f};
    v8f cR = cL;
    {
      Frag16 bk0, bk1, bk2, bk3;
      const unsigned short* kp  = Kt_ + r * 64 + 16 * hf;   // keys 0..15
      const unsigned short* kp2 = kp + 16 * 64;             // keys 16..31
      bk0.q[0] = *(const u32x4*)(kp);       bk0.q[1] = *(const u32x4*)(kp + 8);
      bk1.q[0] = *(const u32x4*)(kp + 32);  bk1.q[1] = *(const u32x4*)(kp + 40);
      bk2.q[0] = *(const u32x4*)(kp2);      bk2.q[1] = *(const u32x4*)(kp2 + 8);
      bk3.q[0] = *(const u32x4*)(kp2 + 32); bk3.q[1] = *(const u32x4*)(kp2 + 40);
      cL = wmma_bf16(aq0.v, bk0.v, cL);
      cL = wmma_bf16(aq1.v, bk1.v, cL);
      cR = wmma_bf16(aq0.v, bk2.v, cR);
      cR = wmma_bf16(aq1.v, bk3.v, cR);
    }

    // online softmax; write P (bf16) into per-wave LDS scratch row-major [16][32]
    unsigned short* pb = &Pb[wave][0] + (8 * hf) * 32 + r;
#pragma unroll
    for (int j = 0; j < 8; ++j) {
      float sl = cL[j] * 0.125f;                 // 1/sqrt(64)
      float sr = cR[j] * 0.125f;
      float mt = red_max16(fmaxf(sl, sr));
      float mnew = fmaxf(mrun[j], mt);
      float al = __expf(mrun[j] - mnew);
      float pl = __expf(sl - mnew);
      float pr = __expf(sr - mnew);
      float rs = red_sum16(pl + pr);
      lrun[j] = lrun[j] * al + rs;
      mrun[j] = mnew;
      o[0][j] *= al; o[1][j] *= al; o[2][j] *= al; o[3][j] *= al;
      pb[j * 32]      = f2bf(pl);
      pb[j * 32 + 16] = f2bf(pr);
    }
    asm volatile("s_wait_dscnt 0" ::: "memory");  // wave-local LDS RAW fence

    // O += P * V : V^T tile rows are exactly the B-fragment chunks
    Frag16 ap;
    const unsigned short* pr_ = &Pb[wave][0] + r * 32;
    ap.q[0] = *(const u32x4*)(pr_ + 16 * hf);
    ap.q[1] = *(const u32x4*)(pr_ + 16 * hf + 8);
    Frag16 bv[4];
#pragma unroll
    for (int f = 0; f < 4; ++f) {
      const unsigned short* vp = Vt_ + (f * 16 + r) * 32 + 16 * hf;
      bv[f].q[0] = *(const u32x4*)vp;
      bv[f].q[1] = *(const u32x4*)(vp + 8);
    }
#pragma unroll
    for (int f = 0; f < 4; ++f) o[f] = wmma_bf16(ap.v, bv[f].v, o[f]);

    __syncthreads();  // all waves done with buf before it is overwritten
  }

  // epilogue: normalize and store bf16 attn output [row][head*64+hd]
#pragma unroll
  for (int f = 0; f < 4; ++f) {
#pragma unroll
    for (int j = 0; j < 8; ++j) {
      float v = o[f][j] / lrun[j];
      size_t row = rowbase + qb + j + 8 * hf;
      outbf[row * 1024 + head * 64 + f * 16 + r] = f2bf(v);
    }
  }
}

// ---------- host ----------
extern "C" void kernel_launch(void* const* d_in, const int* in_sizes, int n_in,
                              void* d_out, int out_size, void* d_ws, size_t ws_size,
                              hipStream_t stream) {
  (void)in_sizes; (void)n_in; (void)out_size; (void)ws_size;
  const float* x    = (const float*)d_in[0];
  const float* Wqkv = (const float*)d_in[1];
  const float* bqkv = (const float*)d_in[2];
  const float* Wout = (const float*)d_in[3];
  const float* bout = (const float*)d_in[4];

  unsigned short* xbf    = (unsigned short*)d_ws;                 // 8192x1024 bf16 (16MB)
  unsigned short* wqkvT  = xbf   + (size_t)8192 * 1024;           // 3072x1024 bf16 (6MB)
  unsigned short* woutT  = wqkvT + (size_t)3072 * 1024;           // 1024x1024 bf16 (2MB)
  unsigned short* qkvbf  = woutT + (size_t)1024 * 1024;           // 8192x3072 bf16 (48MB)
  unsigned short* vTbf   = qkvbf + (size_t)8192 * 3072;           // 4x1024x2048 bf16 (16MB)
  unsigned short* attnbf = xbf;  // reuse: xbf dead after QKV GEMM

  cast_bf<<<(8192 * 1024) / 256, 256, 0, stream>>>(x, xbf, 8192 * 1024);
  tcast_bf<<<dim3(3072 / 16, 1024 / 16), dim3(16, 16), 0, stream>>>(Wqkv, wqkvT, 1024, 3072);
  tcast_bf<<<dim3(1024 / 16, 1024 / 16), dim3(16, 16), 0, stream>>>(Wout, woutT, 1024, 1024);

  // QKV = x @ W_qkv + b_qkv  (bf16 out)
  gemm_bf16<<<dim3(3072 / 128, 8192 / 128), 256, 0, stream>>>(
      xbf, wqkvT, bqkv, qkvbf, 3072, 1024, 1);

  // V^T for the P@V operand path
  vtrans_bf<<<dim3(2048 / 16, 1024 / 16, 4), dim3(16, 16), 0, stream>>>(qkvbf, vTbf);

  // fused attention (flash-style, TDM-staged K and V^T tiles)
  attn_kernel<<<dim3(SEQ / 128, 16, 4), 256, 0, stream>>>(qkvbf, vTbf, attnbf);

  // out = attn @ W_out + b_out  (fp32 out)
  gemm_bf16<<<dim3(1024 / 128, 8192 / 128), 256, 0, stream>>>(
      attnbf, woutT, bout, d_out, 1024, 1024, 0);
}
